// MultiHeadAttention_50964081934615
// MI455X (gfx1250) — matrix operand order repaired and checked by reading, not compiled, not run on hardware
//
#include <hip/hip_runtime.h>
#include <hip/hip_bf16.h>

// ---------------------------------------------------------------------------
// MHA forward for MI455X (gfx1250, wave32, WMMA bf16 -> f32 accumulate)
// Tile staging uses GLOBAL_LOAD_ASYNC_TO_LDS_B128 (ASYNCcnt) when available.
// ---------------------------------------------------------------------------

typedef __bf16 v16bf __attribute__((ext_vector_type(16)));
typedef float  v8f   __attribute__((ext_vector_type(8)));
typedef int    v4i_  __attribute__((ext_vector_type(4)));

union FragU { uint4 q[2]; v16bf v; };

#if defined(__HIP_DEVICE_COMPILE__) && __has_builtin(__builtin_amdgcn_global_load_async_to_lds_b128)
#define ASYNC_LDS_COPY 1
#else
#define ASYNC_LDS_COPY 0
#endif

__device__ __forceinline__ void copy16_g2lds(const __bf16* __restrict__ g,
                                             __bf16* __restrict__ l) {
#if ASYNC_LDS_COPY
  // async DMA: global -> LDS, no VGPR round-trip, tracked by ASYNCcnt
  __builtin_amdgcn_global_load_async_to_lds_b128(
      (__attribute__((address_space(1))) v4i_*)(void*)g,
      (__attribute__((address_space(3))) v4i_*)(void*)l, 0, 0);
#else
  *(uint4*)l = *(const uint4*)g;
#endif
}

__device__ __forceinline__ void wait_async_copies() {
#if ASYNC_LDS_COPY
#if __has_builtin(__builtin_amdgcn_s_wait_asynccnt)
  __builtin_amdgcn_s_wait_asynccnt(0);
#else
  asm volatile("s_wait_asynccnt 0x0" ::: "memory");
#endif
#endif
}

__device__ __forceinline__ v8f wmma_bf16f32(const v16bf& a, const v16bf& b, const v8f& c) {
  // 8 args: (neg_a, A, neg_b, B, c_mod, C, reuse_a, reuse_b)
  return __builtin_amdgcn_wmma_f32_16x16x32_bf16(false, a, false, b, (short)0, c, false, false);
}

constexpr int kB  = 2;
constexpr int kS  = 2048;
constexpr int kD  = 1024;
constexpr int kH  = 16;
constexpr int kDK = 64;

// ---------------------------------------------------------------------------
// fp32 -> bf16 conversion (vectorized, 4 elems/thread)
// ---------------------------------------------------------------------------
__global__ __launch_bounds__(256)
void cvt_f32_to_bf16(const float* __restrict__ x, __bf16* __restrict__ y, long n4) {
  long i = (long)blockIdx.x * 256 + threadIdx.x;
  if (i >= n4) return;
  const float4 f = ((const float4*)x)[i];
  union { __bf16 h[4]; unsigned long long u; } o;
  o.h[0] = (__bf16)f.x; o.h[1] = (__bf16)f.y;
  o.h[2] = (__bf16)f.z; o.h[3] = (__bf16)f.w;
  ((unsigned long long*)y)[i] = o.u;
}

// ---------------------------------------------------------------------------
// Generic NT GEMM: C[m,n] = alpha * sum_k A[m,k]*B[n,k] + bias[n]
//   A: [M,K] bf16 row-major (lda), batched via sAz
//   B: [N,K] bf16 row-major (ldb), batched via sBz
// OUT_MODE: 0 = bf16 split-heads [B,H,S,dk]
//           1 = bf16 split-heads transposed [B,H,dk,S]
//           2 = fp32 batched row-major (z*sCz + m*ldc + n)
//           3 = bf16 merge-heads [B,S,D] (z = b*H + h)
// ---------------------------------------------------------------------------
template<int BM, int BN, int BK, int OUT_MODE>
__global__ __launch_bounds__(256)
void gemm_bf16_nt(const __bf16* __restrict__ A, int lda, long sAz,
                  const __bf16* __restrict__ Bm, int ldb, long sBz,
                  const float* __restrict__ bias,
                  void* __restrict__ C, long sCz, int ldc,
                  int K, float alpha)
{
  constexpr int LSTR  = BK + 8;         // padded row stride: 144B -> bank-conflict-free
  constexpr int WROWS = 4, WCOLS = 2;   // 8 waves
  constexpr int WM = BM / WROWS, WN = BN / WCOLS;
  constexpr int TM = WM / 16, TN = WN / 16;

  __shared__ alignas(16) __bf16 As[BM * LSTR];
  __shared__ alignas(16) __bf16 Bs[BN * LSTR];

  const int tid  = threadIdx.x;
  const int lane = tid & 31;     // wave32
  const int wid  = tid >> 5;
  const int wm   = wid & 3;
  const int wn   = wid >> 2;
  const int l16  = lane & 15;
  const int lhi  = lane >> 4;
  const int z    = blockIdx.z;

  const __bf16* Ab = A  + (size_t)z * sAz + (size_t)blockIdx.x * BM * lda;
  const __bf16* Bb = Bm + (size_t)z * sBz + (size_t)blockIdx.y * BN * ldb;

  v8f acc[TM][TN];
  #pragma unroll
  for (int i = 0; i < TM; ++i)
    #pragma unroll
    for (int j = 0; j < TN; ++j)
      acc[i][j] = (v8f)0.0f;

  constexpr int PA = (BM * BK) / (256 * 8);
  constexpr int PB = (BN * BK) / (256 * 8);

  for (int k0 = 0; k0 < K; k0 += BK) {
    // stage A tile (BM x BK) and B tile (BN x BK) into LDS, 16B per thread/pass
    #pragma unroll
    for (int p = 0; p < PA; ++p) {
      const int idx = p * 2048 + tid * 8;
      const int r = idx / BK, c = idx % BK;
      copy16_g2lds(Ab + (size_t)r * lda + k0 + c, &As[r * LSTR + c]);
    }
    #pragma unroll
    for (int p = 0; p < PB; ++p) {
      const int idx = p * 2048 + tid * 8;
      const int r = idx / BK, c = idx % BK;
      copy16_g2lds(Bb + (size_t)r * ldb + k0 + c, &Bs[r * LSTR + c]);
    }
    if (k0 + BK < K) {   // hint next K-tile into cache (global_prefetch_b8)
      __builtin_prefetch(Ab + (size_t)(tid >> 1) * lda + k0 + BK, 0, 1);
      __builtin_prefetch(Bb + (size_t)(tid >> 1) * ldb + k0 + BK, 0, 1);
    }
    wait_async_copies();   // drain ASYNCcnt before making tile visible
    __syncthreads();

    #pragma unroll
    for (int kk = 0; kk < BK; kk += 32) {
      FragU a[TM], b[TN];
      // A fragment: 16-bit A 16x32 layout -> lane(m=l16, half=lhi), two b128 reads
      #pragma unroll
      for (int tm = 0; tm < TM; ++tm) {
        const __bf16* base = &As[(wm * WM + tm * 16 + l16) * LSTR + kk];
        a[tm].q[0] = *(const uint4*)(base + lhi * 8);
        a[tm].q[1] = *(const uint4*)(base + 16 + lhi * 8);
      }
      // B fragment: 16-bit B 32x16 layout -> lane(n=l16, half=lhi), two b128 reads
      #pragma unroll
      for (int tn = 0; tn < TN; ++tn) {
        const __bf16* base = &Bs[(wn * WN + tn * 16 + l16) * LSTR + kk];
        b[tn].q[0] = *(const uint4*)(base + lhi * 8);
        b[tn].q[1] = *(const uint4*)(base + 16 + lhi * 8);
      }
      #pragma unroll
      for (int tm = 0; tm < TM; ++tm)
        #pragma unroll
        for (int tn = 0; tn < TN; ++tn)
          acc[tm][tn] = wmma_bf16f32(a[tm].v, b[tn].v, acc[tm][tn]);
    }
    __syncthreads();
  }

  // epilogue: C/D layout: VGPR r -> M = r + 8*lhi, N = l16
  const int gm0 = blockIdx.x * BM + wm * WM;
  const int gn0 = blockIdx.y * BN + wn * WN;
  #pragma unroll
  for (int tm = 0; tm < TM; ++tm) {
    #pragma unroll
    for (int tn = 0; tn < TN; ++tn) {
      const int gn = gn0 + tn * 16 + l16;
      const float bv = bias ? bias[gn] : 0.0f;
      #pragma unroll
      for (int r = 0; r < 8; ++r) {
        const int gm = gm0 + tm * 16 + lhi * 8 + r;
        const float val = acc[tm][tn][r] * alpha + bv;
        if constexpr (OUT_MODE == 0) {
          const int b = gm >> 11, s = gm & 2047, h = gn >> 6, d = gn & 63;
          ((__bf16*)C)[(((size_t)b * kH + h) * kS + s) * kDK + d] = (__bf16)val;
        } else if constexpr (OUT_MODE == 1) {
          const int b = gm >> 11, s = gm & 2047, h = gn >> 6, d = gn & 63;
          ((__bf16*)C)[(((size_t)b * kH + h) * kDK + d) * kS + s] = (__bf16)val;
        } else if constexpr (OUT_MODE == 2) {
          ((float*)C)[(size_t)z * sCz + (size_t)gm * ldc + gn] = val;
        } else {
          const int b = z >> 4, h = z & 15;
          ((__bf16*)C)[((size_t)b * kS + gm) * kD + h * kDK + gn] = (__bf16)val;
        }
      }
    }
  }
}

// ---------------------------------------------------------------------------
// Row softmax over S=2048, in place (fp32) + bf16 copy for the ctx GEMM
// ---------------------------------------------------------------------------
__global__ __launch_bounds__(256)
void softmax_rows_kernel(float* __restrict__ P, __bf16* __restrict__ Pb) {
  __shared__ float red[256];
  const size_t row = blockIdx.x;
  float*  p  = P  + row * (size_t)kS;
  __bf16* pb = Pb + row * (size_t)kS;
  const int t = threadIdx.x;

  float v[8];
  float m = -3.402823466e38f;
  #pragma unroll
  for (int i = 0; i < 8; ++i) { v[i] = p[t + (i << 8)]; m = fmaxf(m, v[i]); }
  red[t] = m; __syncthreads();
  for (int s = 128; s > 0; s >>= 1) { if (t < s) red[t] = fmaxf(red[t], red[t + s]); __syncthreads(); }
  m = red[0]; __syncthreads();

  float sum = 0.0f;
  #pragma unroll
  for (int i = 0; i < 8; ++i) { v[i] = __expf(v[i] - m); sum += v[i]; }
  red[t] = sum; __syncthreads();
  for (int s = 128; s > 0; s >>= 1) { if (t < s) red[t] += red[t + s]; __syncthreads(); }
  const float inv = 1.0f / red[0];

  #pragma unroll
  for (int i = 0; i < 8; ++i) {
    const float r = v[i] * inv;
    p[t + (i << 8)]  = r;
    pb[t + (i << 8)] = (__bf16)r;
  }
}

// ---------------------------------------------------------------------------
// residual + LayerNorm over D=1024 (fp32)
// ---------------------------------------------------------------------------
__global__ __launch_bounds__(256)
void resid_ln_kernel(const float* __restrict__ x, const float* __restrict__ o,
                     const float* __restrict__ g, const float* __restrict__ bta,
                     float* __restrict__ y) {
  __shared__ float r1[256];
  __shared__ float r2[256];
  const size_t row = blockIdx.x;
  const int t = threadIdx.x;
  const float* xr = x + row * (size_t)kD;
  const float* orr = o + row * (size_t)kD;

  float v[4], s1 = 0.0f, s2 = 0.0f;
  #pragma unroll
  for (int i = 0; i < 4; ++i) {
    const int c = t + (i << 8);
    v[i] = xr[c] + orr[c];
    s1 += v[i];
    s2 += v[i] * v[i];
  }
  r1[t] = s1; r2[t] = s2; __syncthreads();
  for (int s = 128; s > 0; s >>= 1) {
    if (t < s) { r1[t] += r1[t + s]; r2[t] += r2[t + s]; }
    __syncthreads();
  }
  const float mu  = r1[0] * (1.0f / kD);
  const float var = r2[0] * (1.0f / kD) - mu * mu;
  const float rs  = rsqrtf(var + 1e-5f);
  #pragma unroll
  for (int i = 0; i < 4; ++i) {
    const int c = t + (i << 8);
    y[row * (size_t)kD + c] = (v[i] - mu) * rs * g[c] + bta[c];
  }
}

// ---------------------------------------------------------------------------
// Host launcher
// ---------------------------------------------------------------------------
extern "C" void kernel_launch(void* const* d_in, const int* in_sizes, int n_in,
                              void* d_out, int out_size, void* d_ws, size_t ws_size,
                              hipStream_t stream) {
  (void)in_sizes; (void)n_in; (void)out_size; (void)ws_size;

  const float* query = (const float*)d_in[0];
  const float* key_  = (const float*)d_in[1];
  const float* value = (const float*)d_in[2];
  const float* w_q = (const float*)d_in[3];
  const float* b_q = (const float*)d_in[4];
  const float* w_k = (const float*)d_in[5];
  const float* b_k = (const float*)d_in[6];
  const float* w_v = (const float*)d_in[7];
  const float* b_v = (const float*)d_in[8];
  const float* w_o = (const float*)d_in[9];
  const float* b_o = (const float*)d_in[10];
  const float* ln_g = (const float*)d_in[11];
  const float* ln_b = (const float*)d_in[12];

  float* y_out    = (float*)d_out;
  float* attn_out = (float*)d_out + (size_t)kB * kS * kD;

  char* ws = (char*)d_ws;
  size_t off = 0;
  auto alloc = [&](size_t bytes) -> void* {
    void* p = ws + off;
    off = (off + bytes + 255) & ~(size_t)255;
    return p;
  };
  const size_t nBSD  = (size_t)kB * kS * kD;       // 4,194,304
  const size_t nDD   = (size_t)kD * kD;            // 1,048,576
  const size_t nAttn = (size_t)kB * kH * kS * kS;  // 134,217,728

  __bf16* xq   = (__bf16*)alloc(nBSD * 2);
  __bf16* xk   = (__bf16*)alloc(nBSD * 2);
  __bf16* xv   = (__bf16*)alloc(nBSD * 2);
  __bf16* wqb  = (__bf16*)alloc(nDD * 2);
  __bf16* wkb  = (__bf16*)alloc(nDD * 2);
  __bf16* wvb  = (__bf16*)alloc(nDD * 2);
  __bf16* wob  = (__bf16*)alloc(nDD * 2);
  __bf16* Qb   = (__bf16*)alloc(nBSD * 2);   // [B,H,S,dk]
  __bf16* Kb   = (__bf16*)alloc(nBSD * 2);   // [B,H,S,dk]
  __bf16* Vt   = (__bf16*)alloc(nBSD * 2);   // [B,H,dk,S]  (transposed for NT ctx GEMM)
  __bf16* Pb   = (__bf16*)alloc(nAttn * 2);  // bf16 softmax probs
  __bf16* ctxb = (__bf16*)alloc(nBSD * 2);   // [B,S,D]
  float*  oprj = (float*) alloc(nBSD * 4);   // out-projection fp32

  // fp32 -> bf16 conversions
  cvt_f32_to_bf16<<<dim3((unsigned)(nBSD / 1024)), 256, 0, stream>>>(query, xq, (long)(nBSD / 4));
  cvt_f32_to_bf16<<<dim3((unsigned)(nBSD / 1024)), 256, 0, stream>>>(key_,  xk, (long)(nBSD / 4));
  cvt_f32_to_bf16<<<dim3((unsigned)(nBSD / 1024)), 256, 0, stream>>>(value, xv, (long)(nBSD / 4));
  cvt_f32_to_bf16<<<dim3((unsigned)(nDD / 1024)), 256, 0, stream>>>(w_q, wqb, (long)(nDD / 4));
  cvt_f32_to_bf16<<<dim3((unsigned)(nDD / 1024)), 256, 0, stream>>>(w_k, wkb, (long)(nDD / 4));
  cvt_f32_to_bf16<<<dim3((unsigned)(nDD / 1024)), 256, 0, stream>>>(w_v, wvb, (long)(nDD / 4));
  cvt_f32_to_bf16<<<dim3((unsigned)(nDD / 1024)), 256, 0, stream>>>(w_o, wob, (long)(nDD / 4));

  const dim3 blk(256);

  // Q/K/V projections: [4096,1024] = x @ W^T + b  (NT GEMM)
  gemm_bf16_nt<128,128,64,0><<<dim3(32,8,1), blk, 0, stream>>>(
      xq, kD, 0, wqb, kD, 0, b_q, Qb, 0, 0, kD, 1.0f);
  gemm_bf16_nt<128,128,64,0><<<dim3(32,8,1), blk, 0, stream>>>(
      xk, kD, 0, wkb, kD, 0, b_k, Kb, 0, 0, kD, 1.0f);
  gemm_bf16_nt<128,128,64,1><<<dim3(32,8,1), blk, 0, stream>>>(
      xv, kD, 0, wvb, kD, 0, b_v, Vt, 0, 0, kD, 1.0f);

  // scores = Q @ K^T / sqrt(dk), batched over B*H, written into attn region
  gemm_bf16_nt<128,128,64,2><<<dim3(16,16,kB*kH), blk, 0, stream>>>(
      Qb, kDK, (long)kS * kDK, Kb, kDK, (long)kS * kDK, nullptr,
      attn_out, (long)kS * kS, kS, kDK, 0.125f);

  // softmax in place (fp32) + bf16 copy
  softmax_rows_kernel<<<dim3(kB * kH * kS), blk, 0, stream>>>(attn_out, Pb);

  // ctx = P @ V, batched; Vt is [dk,S] so this is NT with K = S
  gemm_bf16_nt<128,64,64,3><<<dim3(16,1,kB*kH), blk, 0, stream>>>(
      Pb, kS, (long)kS * kS, Vt, kS, (long)kDK * kS, nullptr,
      ctxb, 0, 0, kS, 1.0f);

  // out projection: [4096,1024] = ctx @ Wo^T + b_o  (fp32 output for LN)
  gemm_bf16_nt<128,128,64,2><<<dim3(32,8,1), blk, 0, stream>>>(
      ctxb, kD, 0, wob, kD, 0, b_o, oprj, 0, kD, kD, 1.0f);

  // y = LayerNorm(query + out) * gamma + beta
  resid_ln_kernel<<<dim3(kB * kS), blk, 0, stream>>>(query, oprj, ln_g, ln_b, y_out);
}
